// AlibiAttentionWrapper_70463233458701
// MI455X (gfx1250) — matrix operand order repaired and checked by reading, not compiled, not run on hardware
//
#include <hip/hip_runtime.h>
#include <cstdint>

typedef __bf16 bf16_t;
typedef bf16_t v16bf __attribute__((ext_vector_type(16)));
typedef float  v8f   __attribute__((ext_vector_type(8)));

#if defined(__has_builtin)
# if __has_builtin(__builtin_amdgcn_global_load_async_to_lds_b128)
#  define HAVE_ASYNC 1
# endif
#endif
#ifndef HAVE_ASYNC
# define HAVE_ASYNC 0
#endif

#define AS1 __attribute__((address_space(1)))
#define AS3 __attribute__((address_space(3)))

union FragU { v16bf v; uint32_t u[8]; };

// ---------------------------------------------------------------------------
// WMMA fragment helpers (bf16, 16x16x32). Layouts per CDNA5 ISA 7.12.2:
//  A (16Mx32K):  lane m = lane&15, half = lane>>4; VGPR v holds
//                k = (v>>2)*16 + half*8 + (v&3)*2  (+0,+1 packed)
//  B (32Kx16N):  source tile stored as [n][k] (k contiguous); lane n = lane&15,
//                half = lane>>4; VGPR v holds k = (v>>2)*16 + half*8 + (v&3)*2 (+0,+1 packed)
//  C/D (16x16 f32): VGPR r, lane l -> M = r + 8*(l>>4), N = l&15
// ---------------------------------------------------------------------------
__device__ __forceinline__ v16bf load_frag_a(const bf16_t* p, int ld, int lane) {
  const int m = lane & 15, hh = lane >> 4;
  FragU f;
#pragma unroll
  for (int v = 0; v < 8; ++v) {
    const int k = ((v >> 2) << 4) + (hh << 3) + ((v & 3) << 1);
    f.u[v] = *(const uint32_t*)(p + (size_t)m * ld + k);
  }
  return f.v;
}

__device__ __forceinline__ v16bf load_frag_b(const bf16_t* p, int ld, int lane) {
  const int n = lane & 15, hh = lane >> 4;
  FragU f;
#pragma unroll
  for (int v = 0; v < 8; ++v) {
    const int k = ((v >> 2) << 4) + (hh << 3) + ((v & 3) << 1);
    f.u[v] = *(const uint32_t*)(p + (size_t)n * ld + k);
  }
  return f.v;
}

__device__ __forceinline__ v8f wmma_bf16(v16bf a, v16bf b, v8f c) {
  // D = A x B + C, f32 accumulate (v_wmma_f32_16x16x32_bf16)
  return __builtin_amdgcn_wmma_f32_16x16x32_bf16(
      /*neg_a=*/false, a, /*neg_b=*/false, b,
      /*c_mod=*/(short)0, c, /*reuse_a=*/false, /*reuse_b=*/false);
}

// ---------------------------------------------------------------------------
// Async global->LDS copy of 16 bf16 (two B128 transfers), ASYNCcnt-tracked.
// Both src and dst must be 16B aligned. Falls back to direct b128 copies.
// Builtin signature (from compiler diagnostic): param 1 is
// `int4 __device__*` (AS1 global), param 2 is the LDS side (AS3), then
// imm offset + imm cpol.
// ---------------------------------------------------------------------------
typedef __attribute__((__vector_size__(4 * sizeof(int)))) int v4i;

#if HAVE_ASYNC
__device__ __forceinline__ AS1 v4i* to_global_v4(const void* p) {
  return (AS1 v4i*)(uintptr_t)p;
}
__device__ __forceinline__ AS3 v4i* to_lds_v4(void* p) {
  return (AS3 v4i*)(uint32_t)(uintptr_t)p;  // low 32 bits = LDS offset
}
#endif

__device__ __forceinline__ void copy16_g2l(const bf16_t* src, bf16_t* dst) {
#if HAVE_ASYNC
  AS1 v4i* gs = to_global_v4(src);
  AS3 v4i* ls = to_lds_v4(dst);
  __builtin_amdgcn_global_load_async_to_lds_b128(gs, ls, 0, 0);
  __builtin_amdgcn_global_load_async_to_lds_b128(gs + 1, ls + 1, 0, 0);
#else
  uint4 d0 = *(const uint4*)src;
  uint4 d1 = *(const uint4*)(src + 8);
  *(uint4*)dst = d0;
  *(uint4*)(dst + 8) = d1;
#endif
}

__device__ __forceinline__ void wait_async0() {
#if HAVE_ASYNC
# if defined(__has_builtin) && __has_builtin(__builtin_amdgcn_s_wait_asynccnt)
  __builtin_amdgcn_s_wait_asynccnt(0);
# else
  asm volatile("s_wait_asynccnt 0" ::: "memory");
# endif
#endif
}

// ---------------------------------------------------------------------------
// fp32 -> bf16 conversion (grid-stride)
// ---------------------------------------------------------------------------
__global__ void f32_to_bf16_kernel(const float* __restrict__ in,
                                   bf16_t* __restrict__ out, int n) {
  int i = blockIdx.x * blockDim.x + threadIdx.x;
  const int stride = gridDim.x * blockDim.x;
  for (; i < n; i += stride) out[i] = (bf16_t)in[i];
}

// ---------------------------------------------------------------------------
// C[M,N] = A[M,K](bf16) @ W[N,K](bf16)^T + bias[N](f32)
// Block: 256 threads = 8 waves, 4(M) x 2(N) wave grid; block tile 128x128,
// wave tile 32x64 (2x4 WMMA tiles), BK = 32.
// Double-buffered LDS fed by GLOBAL_LOAD_ASYNC_TO_LDS_B128 (ASYNCcnt) when
// available. LDS rows padded to 40 elems: 80B rows (16B aligned for B128),
// 20-dword stride is bank-conflict-free across the 16 row-lanes.
// ---------------------------------------------------------------------------
template <typename OutT>
__global__ __launch_bounds__(256) void gemm_bias_kernel(
    const bf16_t* __restrict__ A, const bf16_t* __restrict__ W,
    const float* __restrict__ bias, OutT* __restrict__ C,
    int M, int N, int K) {
  __shared__ bf16_t sA[2][128 * 40];
  __shared__ bf16_t sB[2][128 * 40];

  const int t = threadIdx.x;
  const int lane = t & 31;
  const int wave = t >> 5;
  const int wm = wave & 3;   // wave M index (0..3) -> 32 rows each
  const int wn = wave >> 2;  // wave N index (0..1) -> 64 cols each
  const int bm = blockIdx.y * 128;
  const int bn = blockIdx.x * 128;

  // staging map: 128 rows x 32 k, 16 elems (two B128) per thread
  const int srow = t >> 1, sks = (t & 1) << 4;
  const bf16_t* gA = A + (size_t)(bm + srow) * K + sks;
  const bf16_t* gB = W + (size_t)(bn + srow) * K + sks;

  v8f acc[2][4] = {};

  int buf = 0;
  copy16_g2l(gA, &sA[0][srow * 40 + sks]);
  copy16_g2l(gB, &sB[0][srow * 40 + sks]);

  for (int k0 = 0; k0 < K; k0 += 32) {
    wait_async0();
    __syncthreads();

    if (k0 + 32 < K) {  // prefetch next K-slab into the other buffer
      copy16_g2l(gA + k0 + 32, &sA[buf ^ 1][srow * 40 + sks]);
      copy16_g2l(gB + k0 + 32, &sB[buf ^ 1][srow * 40 + sks]);
    }

    v16bf af[2], bfr[4];
    af[0] = load_frag_a(&sA[buf][(wm * 32 + 0) * 40], 40, lane);
    af[1] = load_frag_a(&sA[buf][(wm * 32 + 16) * 40], 40, lane);
#pragma unroll
    for (int nt = 0; nt < 4; ++nt)
      bfr[nt] = load_frag_b(&sB[buf][(wn * 64 + nt * 16) * 40], 40, lane);
#pragma unroll
    for (int mt = 0; mt < 2; ++mt)
#pragma unroll
      for (int nt = 0; nt < 4; ++nt)
        acc[mt][nt] = wmma_bf16(af[mt], bfr[nt], acc[mt][nt]);

    buf ^= 1;
  }

  const int hh = lane >> 4, nl = lane & 15;
#pragma unroll
  for (int mt = 0; mt < 2; ++mt) {
#pragma unroll
    for (int nt = 0; nt < 4; ++nt) {
      const int ncol = bn + wn * 64 + nt * 16 + nl;
      const float bv = bias[ncol];
#pragma unroll
      for (int r = 0; r < 8; ++r) {
        const int mrow = bm + wm * 32 + mt * 16 + r + (hh << 3);
        C[(size_t)mrow * N + ncol] = (OutT)(acc[mt][nt][r] + bv);
      }
    }
  }
}

// ---------------------------------------------------------------------------
// Flash attention with ALiBi bias.
// qkv: [B*S, 3072] bf16, cols 0..1023=Q, 1024..2047=K, 2048..3071=V (h*64+d)
// attn: [B*S, 1024] bf16 in [B,S,H*D] layout.
// Block = 8 waves; wave owns a 16-query tile; block covers 128 queries of one
// (b,h). Key blocks of 64 staged in LDS (K via async B128 when available,
// V transposed manually). Per key block: 8 WMMAs QK^T, online softmax,
// P->LDS->A-fragment, 8 WMMAs PV.
// ---------------------------------------------------------------------------
__global__ __launch_bounds__(256) void alibi_flash_attn_kernel(
    const bf16_t* __restrict__ qkv, bf16_t* __restrict__ attn) {
  __shared__ bf16_t sK[64 * 72];       // [key][dim], 144B rows (16B aligned)
  __shared__ bf16_t sV[64 * 66];       // transposed: [dim][key]
  __shared__ bf16_t sP[8 * 16 * 66];   // per-wave P scratch [qrow][key]

  const int t = threadIdx.x;
  const int lane = t & 31;
  const int wave = t >> 5;
  const int bh = blockIdx.y;
  const int b = bh >> 4;
  const int h = bh & 15;
  const int q0 = blockIdx.x * 128 + wave * 16;  // first query row of this wave
  const float slope = exp2f(-0.5f * (float)(h + 1));
  const int hh = lane >> 4;
  const int nl = lane & 15;

  // Q fragments: rows q0..q0+15, dims 0..63 (two K=32 chunks)
  const bf16_t* qbase = qkv + (size_t)(b * 2048 + q0) * 3072 + h * 64;
  const v16bf qa0 = load_frag_a(qbase, 3072, lane);
  const v16bf qa1 = load_frag_a(qbase + 32, 3072, lane);

  v8f o[4] = {};
  float m[8], l[8];
#pragma unroll
  for (int r = 0; r < 8; ++r) { m[r] = -1e30f; l[r] = 0.0f; }

  for (int kb = 0; kb < 2048; kb += 64) {
    __syncthreads();
    {  // stage K (async) and V (transposed): thread -> (key, 16-dim segment)
      const int key = t >> 2;
      const int ds0 = (t & 3) << 4;
      const bf16_t* kr = qkv + (size_t)(b * 2048 + kb + key) * 3072 + 1024 + h * 64 + ds0;
      const bf16_t* vr = kr + 1024;
      copy16_g2l(kr, &sK[key * 72 + ds0]);
      union { uint4 q; uint32_t w[4]; bf16_t e[8]; } v0, v1;
      v0.q = *(const uint4*)vr;
      v1.q = *(const uint4*)(vr + 8);
#pragma unroll
      for (int e = 0; e < 8; ++e) {
        sV[(ds0 + e) * 66 + key]     = v0.e[e];
        sV[(ds0 + 8 + e) * 66 + key] = v1.e[e];
      }
    }
    wait_async0();
    __syncthreads();

    // scores: 4 key sub-tiles x (K=64 as 2 chained WMMAs)
    v8f sv[4];
#pragma unroll
    for (int nt = 0; nt < 4; ++nt) {
      v8f s = {};
      const v16bf kb0 = load_frag_b(&sK[(nt * 16) * 72], 72, lane);       // dims 0..31
      const v16bf kb1 = load_frag_b(&sK[(nt * 16) * 72 + 32], 72, lane);  // dims 32..63
      s = wmma_bf16(qa0, kb0, s);
      s = wmma_bf16(qa1, kb1, s);
      sv[nt] = s;
    }

    // scale + ALiBi bias + online softmax (row reductions across 16-lane groups)
#pragma unroll
    for (int r = 0; r < 8; ++r) {
      const int i = q0 + r + (hh << 3);
      float mx = -1e30f;
#pragma unroll
      for (int nt = 0; nt < 4; ++nt) {
        const int j = kb + nt * 16 + nl;
        const float val = sv[nt][r] * 0.125f + fminf(slope * (float)(i - j), 8.0f);
        sv[nt][r] = val;
        mx = fmaxf(mx, val);
      }
      mx = fmaxf(mx, __shfl_xor(mx, 1, 32));
      mx = fmaxf(mx, __shfl_xor(mx, 2, 32));
      mx = fmaxf(mx, __shfl_xor(mx, 4, 32));
      mx = fmaxf(mx, __shfl_xor(mx, 8, 32));
      const float mn = fmaxf(m[r], mx);
      const float al = __expf(m[r] - mn);
      float rs = 0.0f;
#pragma unroll
      for (int nt = 0; nt < 4; ++nt) {
        const float p = __expf(sv[nt][r] - mn);
        sv[nt][r] = p;
        rs += p;
      }
      rs += __shfl_xor(rs, 1, 32);
      rs += __shfl_xor(rs, 2, 32);
      rs += __shfl_xor(rs, 4, 32);
      rs += __shfl_xor(rs, 8, 32);
      l[r] = l[r] * al + rs;
      m[r] = mn;
#pragma unroll
      for (int dt = 0; dt < 4; ++dt) o[dt][r] *= al;
    }

    // P -> per-wave LDS scratch in [row][key] order, then reload as A-fragments
    bf16_t* pw = &sP[wave * 16 * 66];
#pragma unroll
    for (int nt = 0; nt < 4; ++nt)
#pragma unroll
      for (int r = 0; r < 8; ++r)
        pw[(r + (hh << 3)) * 66 + nt * 16 + nl] = (bf16_t)sv[nt][r];

    // PV: two 32-key chunks x four 16-dim output tiles
#pragma unroll
    for (int kc = 0; kc < 2; ++kc) {
      const v16bf pa = load_frag_a(pw + kc * 32, 66, lane);
#pragma unroll
      for (int dt = 0; dt < 4; ++dt) {
        const v16bf vb = load_frag_b(&sV[(dt * 16) * 66 + kc * 32], 66, lane);
        o[dt] = wmma_bf16(pa, vb, o[dt]);
      }
    }
  }

  // normalize and store to attn[B*S, H*D]
#pragma unroll
  for (int dt = 0; dt < 4; ++dt) {
#pragma unroll
    for (int r = 0; r < 8; ++r) {
      const int srow = q0 + r + (hh << 3);
      const int col = h * 64 + dt * 16 + nl;
      attn[(size_t)(b * 2048 + srow) * 1024 + col] = (bf16_t)(o[dt][r] / l[r]);
    }
  }
}

// ---------------------------------------------------------------------------
extern "C" void kernel_launch(void* const* d_in, const int* in_sizes, int n_in,
                              void* d_out, int out_size, void* d_ws, size_t ws_size,
                              hipStream_t stream) {
  (void)in_sizes; (void)n_in; (void)out_size; (void)ws_size;

  const float* x     = (const float*)d_in[0];  // [2,2048,1024]
  const float* w_in  = (const float*)d_in[1];  // [3072,1024]
  const float* b_in  = (const float*)d_in[2];  // [3072]
  const float* w_out = (const float*)d_in[3];  // [1024,1024]
  const float* b_out = (const float*)d_in[4];  // [1024]
  float* out = (float*)d_out;                  // [2,2048,1024]

  // workspace carving (bf16 buffers), total = 48 MiB
  bf16_t* xb    = (bf16_t*)d_ws;             // 4096*1024
  bf16_t* wqb   = xb    + 4096 * 1024;       // 3072*1024
  bf16_t* wob   = wqb   + 3072 * 1024;       // 1024*1024
  bf16_t* qkvb  = wob   + 1024 * 1024;       // 4096*3072
  bf16_t* attnb = qkvb  + (size_t)4096 * 3072;  // 4096*1024

  f32_to_bf16_kernel<<<512, 256, 0, stream>>>(x,     xb,  4096 * 1024);
  f32_to_bf16_kernel<<<512, 256, 0, stream>>>(w_in,  wqb, 3072 * 1024);
  f32_to_bf16_kernel<<<512, 256, 0, stream>>>(w_out, wob, 1024 * 1024);

  // qkv = x @ W_in^T + b_in : M=4096, N=3072, K=1024 -> bf16
  gemm_bias_kernel<bf16_t><<<dim3(3072 / 128, 4096 / 128), 256, 0, stream>>>(
      xb, wqb, b_in, qkvb, 4096, 3072, 1024);

  // attention: grid (S/128, B*H)
  alibi_flash_attn_kernel<<<dim3(2048 / 128, 2 * 16), 256, 0, stream>>>(qkvb, attnb);

  // out = attn @ W_out^T + b_out : M=4096, N=1024, K=1024 -> f32
  gemm_bias_kernel<float><<<dim3(1024 / 128, 4096 / 128), 256, 0, stream>>>(
      attnb, wob, b_out, out, 4096, 1024, 1024);
}
